// Quantizer_6141803233719
// MI455X (gfx1250) — matrix operand order repaired and checked by reading, not compiled, not run on hardware
//
#include <hip/hip_runtime.h>

// ---------------------------------------------------------------------------
// VQ quantizer for MI455X (gfx1250): fused bf16-WMMA GEMM + argmin.
// z: [8,2048,512] f32 -> 16384 rows; codebook: [8192,512] f32.
// Outputs (flat): [0] loss, [1..1+8388608) z_q_st, then 16384 idx (as float).
// ---------------------------------------------------------------------------

typedef __attribute__((ext_vector_type(16))) __bf16 v16bf;
typedef __attribute__((ext_vector_type(8)))  float  v8f;

union BFr {
    v16bf          v;
    unsigned short us[16];
    uint4          q[2];
};
static_assert(sizeof(BFr) == 32, "bf16 fragment must be 32 bytes");

#define NROWS   16384      // B*T
#define NCODES  8192       // V
#define NC      512        // C (K dimension)
#define ZQ_OFF  1
#define IDX_OFF (1 + NROWS * NC)

__device__ __forceinline__ unsigned short f2bfu(float f) {
    unsigned int u = __float_as_uint(f);
    u += 0x7FFFu + ((u >> 16) & 1u);          // round-to-nearest-even
    return (unsigned short)(u >> 16);
}

// ---------------------------------------------------------------------------
// Kernel 1: repack codebook f32 -> bf16 in WMMA-B fragment order.
// Fragment (nt, kc): per ISA B layout, lane l holds column N = l&15, and half j at
// K = kc*32 + 8*(l>>4) + j (j<8) or +j+8 (j>=8). Stored contiguously so the
// GEMM reads 32 coalesced bytes per lane per fragment.
// ---------------------------------------------------------------------------
__global__ void cb_pack_kernel(const float* __restrict__ cb,
                               unsigned short* __restrict__ ws_cb) {
    const int frag = (blockIdx.x * blockDim.x + threadIdx.x) >> 5; // 8192 frags
    const int lane = threadIdx.x & 31;
    const int nt   = frag >> 4;            // 512 N tiles of 16 codes
    const int kc   = frag & 15;            // 16 K chunks of 32
    const int kofs = kc * 32 + 8 * (lane >> 4);
    BFr t;
    #pragma unroll
    for (int j = 0; j < 16; ++j)
        t.us[j] = f2bfu(cb[(size_t)(nt * 16 + (lane & 15)) * NC + kofs + j + (j & 8)]);
    uint4* dst = (uint4*)ws_cb + ((size_t)frag * 32 + lane) * 2;
    dst[0] = t.q[0];
    dst[1] = t.q[1];
}

// ---------------------------------------------------------------------------
// Kernel 2: per-code squared norms (one wave per code row).
// ---------------------------------------------------------------------------
__global__ void enorm_kernel(const float* __restrict__ cb,
                             float* __restrict__ enorm) {
    const int row  = blockIdx.x * 8 + (threadIdx.x >> 5);
    const int lane = threadIdx.x & 31;
    const float4* p = (const float4*)(cb + (size_t)row * NC);
    float s = 0.f;
    #pragma unroll
    for (int i = 0; i < 4; ++i) {
        float4 v = p[lane + 32 * i];
        s += v.x * v.x + v.y * v.y + v.z * v.z + v.w * v.w;
    }
    #pragma unroll
    for (int m = 16; m >= 1; m >>= 1) s += __shfl_xor(s, m, 32);
    if (lane == 0) enorm[row] = s;
}

// ---------------------------------------------------------------------------
// Kernel 3: fused GEMM(-2 z.e^T) + ||e||^2 + argmin, bf16 WMMA f32-accumulate.
// Block = 8 waves; waves (g,s): row group g (16 rows), N split s (2048 codes).
// Each wave hoists its 16x512 A operand into 16 v16bf fragments (registers),
// then streams 128 N tiles x 16 v_wmma_f32_16x16x32_bf16.
// ---------------------------------------------------------------------------
__launch_bounds__(256, 1)
__global__ void vq_argmin_kernel(const float* __restrict__ z,
                                 const unsigned short* __restrict__ ws_cb,
                                 const float* __restrict__ enorm,
                                 int* __restrict__ ws_idx,
                                 float* __restrict__ out_idx_f) {
    __shared__ float lval[32][4];
    __shared__ int   lidx[32][4];

    const int tid  = threadIdx.x;
    const int lane = tid & 31;
    const int w    = tid >> 5;
    const int g    = w >> 2;                  // row group 0/1
    const int s    = w & 3;                   // N split 0..3
    const int m0   = blockIdx.x * 32 + g * 16;

    // ---- hoist A fragments: row = m0 + (lane&15); K chunks of 32 ----
    // ISA A layout: lane halves [0..8) at K = 8*(l>>4)+j, [8..16) at K+16.
    v16bf a[16];
    {
        const float* zr   = z + (size_t)(m0 + (lane & 15)) * NC;
        const int    bofs = (lane >> 4) * 8;
        #pragma unroll
        for (int f = 0; f < 16; ++f) {
            const int kb = f * 32 + bofs;
            float4 x0 = *(const float4*)(zr + kb);
            float4 x1 = *(const float4*)(zr + kb + 4);
            float4 x2 = *(const float4*)(zr + kb + 16);
            float4 x3 = *(const float4*)(zr + kb + 20);
            BFr t;
            t.us[0] = f2bfu(x0.x); t.us[1] = f2bfu(x0.y);
            t.us[2] = f2bfu(x0.z); t.us[3] = f2bfu(x0.w);
            t.us[4] = f2bfu(x1.x); t.us[5] = f2bfu(x1.y);
            t.us[6] = f2bfu(x1.z); t.us[7] = f2bfu(x1.w);
            t.us[8] = f2bfu(x2.x); t.us[9] = f2bfu(x2.y);
            t.us[10] = f2bfu(x2.z); t.us[11] = f2bfu(x2.w);
            t.us[12] = f2bfu(x3.x); t.us[13] = f2bfu(x3.y);
            t.us[14] = f2bfu(x3.z); t.us[15] = f2bfu(x3.w);
            a[f] = t.v;
        }
    }

    float best[8];
    int   bidx[8];
    #pragma unroll
    for (int v = 0; v < 8; ++v) { best[v] = 3.4e38f; bidx[v] = 0; }

    const int ncol = lane & 15;
    const int nt0  = s * 128;                 // 128 tiles * 16 codes = 2048

    for (int nt = nt0; nt < nt0 + 128; ++nt) {
        v8f c = {0.f, 0.f, 0.f, 0.f, 0.f, 0.f, 0.f, 0.f};
        const uint4* btile = (const uint4*)ws_cb + ((size_t)nt * 16 * 64 + lane * 2);
        #pragma unroll
        for (int kc = 0; kc < 16; ++kc) {
            BFr b;
            b.q[0] = btile[(size_t)kc * 64];
            b.q[1] = btile[(size_t)kc * 64 + 1];
            c = __builtin_amdgcn_wmma_f32_16x16x32_bf16(
                    false, a[kc], false, b.v, (short)0, c, false, false);
        }
        const float en   = enorm[nt * 16 + ncol];
        const int   cand = nt * 16 + ncol;
        #pragma unroll
        for (int v = 0; v < 8; ++v) {
            float d  = __builtin_fmaf(-2.0f, c[v], en); // ||e||^2 - 2 z.e
            bool  lt = d < best[v];
            best[v] = lt ? d : best[v];
            bidx[v] = lt ? cand : bidx[v];
        }
    }

    // cross-lane argmin within each 16-lane C/D group (bit 4 untouched)
    #pragma unroll
    for (int m = 1; m <= 8; m <<= 1) {
        #pragma unroll
        for (int v = 0; v < 8; ++v) {
            float ov = __shfl_xor(best[v], m, 32);
            int   oi = __shfl_xor(bidx[v], m, 32);
            bool take = (ov < best[v]) || (ov == best[v] && oi < bidx[v]);
            best[v] = take ? ov : best[v];
            bidx[v] = take ? oi : bidx[v];
        }
    }

    if ((lane & 15) == 0) {
        #pragma unroll
        for (int v = 0; v < 8; ++v) {
            int r = g * 16 + v + 8 * (lane >> 4);   // row within block (0..31)
            lval[r][s] = best[v];
            lidx[r][s] = bidx[v];
        }
    }
    __syncthreads();

    if (tid < 32) {
        float bv = lval[tid][0];
        int   bi = lidx[tid][0];
        #pragma unroll
        for (int q = 1; q < 4; ++q) {
            float v2 = lval[tid][q];
            int   i2 = lidx[tid][q];
            bool take = (v2 < bv) || (v2 == bv && i2 < bi);
            bv = take ? v2 : bv;
            bi = take ? i2 : bi;
        }
        const int grow = blockIdx.x * 32 + tid;
        ws_idx[grow]    = bi;
        out_idx_f[grow] = (float)bi;
    }
}

// ---------------------------------------------------------------------------
// Kernel 4: gather z_q = e[idx] (== straight-through output), partial loss.
// Block handles 4 rows; fixed-order tree reduction -> deterministic.
// ---------------------------------------------------------------------------
__global__ void gather_loss_kernel(const float* __restrict__ z,
                                   const float* __restrict__ cb,
                                   const int* __restrict__ ws_idx,
                                   float* __restrict__ zq_out,
                                   float* __restrict__ partials) {
    __shared__ float red[256];
    const int t   = threadIdx.x;
    const int row = blockIdx.x * 4 + (t >> 6);
    const int c0  = (t & 63) * 8;
    const int idx = ws_idx[row];

    const float4* zp = (const float4*)(z  + (size_t)row * NC + c0);
    const float4* cp = (const float4*)(cb + (size_t)idx * NC + c0);
    float4*       op = (float4*)(zq_out + (size_t)row * NC + c0);

    float s = 0.f;
    #pragma unroll
    for (int i = 0; i < 2; ++i) {
        float4 zv = zp[i];
        float4 cv = cp[i];
        op[i] = cv;
        float dx = zv.x - cv.x, dy = zv.y - cv.y;
        float dz = zv.z - cv.z, dw = zv.w - cv.w;
        s += dx * dx + dy * dy + dz * dz + dw * dw;
    }
    red[t] = s;
    __syncthreads();
    for (int off = 128; off >= 1; off >>= 1) {
        if (t < off) red[t] += red[t + off];
        __syncthreads();
    }
    if (t == 0) partials[blockIdx.x] = red[0];
}

__global__ void finalize_loss_kernel(const float* __restrict__ partials,
                                     float* __restrict__ loss) {
    __shared__ float red[256];
    const int t = threadIdx.x;
    float s = 0.f;
    for (int i = t; i < 4096; i += 256) s += partials[i];
    red[t] = s;
    __syncthreads();
    for (int off = 128; off >= 1; off >>= 1) {
        if (t < off) red[t] += red[t + off];
        __syncthreads();
    }
    if (t == 0) loss[0] = red[0] * (1.0f / ((float)NROWS * (float)NC)); // BETA=1
}

// ---------------------------------------------------------------------------
extern "C" void kernel_launch(void* const* d_in, const int* in_sizes, int n_in,
                              void* d_out, int out_size, void* d_ws, size_t ws_size,
                              hipStream_t stream) {
    const float* z  = (const float*)d_in[0];   // 8*2048*512 f32
    const float* cb = (const float*)d_in[1];   // 8192*512 f32
    float* out = (float*)d_out;

    // workspace layout (~8.5 MB)
    char* ws = (char*)d_ws;
    unsigned short* ws_cb    = (unsigned short*)(ws);                   // 8,388,608 B
    float*          ws_enorm = (float*)(ws + 8388608);                  //    32,768 B
    int*            ws_idx   = (int*)  (ws + 8388608 + 32768);          //    65,536 B
    float*          ws_part  = (float*)(ws + 8388608 + 32768 + 65536);  //    16,384 B

    cb_pack_kernel      <<<1024, 256, 0, stream>>>(cb, ws_cb);
    enorm_kernel        <<<1024, 256, 0, stream>>>(cb, ws_enorm);
    vq_argmin_kernel    <<< 512, 256, 0, stream>>>(z, ws_cb, ws_enorm,
                                                   ws_idx, out + IDX_OFF);
    gather_loss_kernel  <<<4096, 256, 0, stream>>>(z, cb, ws_idx,
                                                   out + ZQ_OFF, ws_part);
    finalize_loss_kernel<<<   1, 256, 0, stream>>>(ws_part, out);
}
